// MultiHeadAttention_31851477467366
// MI455X (gfx1250) — compile-verified, matrix operand order checked by reading, not run on hardware
//
#include <hip/hip_runtime.h>
#include <hip/hip_bf16.h>

typedef __attribute__((ext_vector_type(16))) _Float16 v16h;
typedef __attribute__((ext_vector_type(8)))  _Float16 v8h;
typedef __attribute__((ext_vector_type(8)))  float    v8f;

#define N_TOK 2925
#define N_PAD 2944
#define DM    1536
#define NH    12
#define HD    128
#define DQKV  4608
#define FF    13
#define HHH   15
#define WWW   15

__device__ __forceinline__ v8f vzero8() {
  v8f z = {0.f,0.f,0.f,0.f,0.f,0.f,0.f,0.f};
  return z;
}

__device__ __forceinline__ v8f wmma_f16(v16h a, v16h b, v8f c) {
  return __builtin_amdgcn_wmma_f32_16x16x32_f16(false, a, false, b, (short)0, c, false, false);
}

// Load a 16x32 f16 fragment (A from row-major MxK, or B from N-major NxK) out of LDS.
// ISA 7.12.2 layout: lane L holds row (L&15); halves i<8 -> k = k0 + (L>>4)*8 + i,
// halves i>=8 -> k = k0 + 16 + (L>>4)*8 + (i-8).  Two 16B LDS loads per fragment.
__device__ __forceinline__ v16h lds_frag(const _Float16* s, int stride, int row0, int k0) {
  int lane = threadIdx.x & 31;
  const _Float16* p = s + (size_t)(row0 + (lane & 15)) * stride + k0 + ((lane >> 4) << 3);
  v8h lo = *(const v8h*)p;
  v8h hi = *(const v8h*)(p + 16);
  v16h f;
#pragma unroll
  for (int i = 0; i < 8; ++i) { f[i] = lo[i]; f[i + 8] = hi[i]; }
  return f;
}

// ---------------- prep kernels ----------------

__global__ __launch_bounds__(256) void pack_x_kernel(const float* __restrict__ x,
                                                     _Float16* __restrict__ xh) {
  int idx = blockIdx.x * 256 + threadIdx.x;
  if (idx >= N_PAD * DM) return;
  int row = idx / DM;
  xh[idx] = (_Float16)((row < N_TOK) ? x[idx] : 0.f);
}

__global__ __launch_bounds__(256) void pack_wqkv_kernel(const float* __restrict__ Wq,
                                                        const float* __restrict__ Wk,
                                                        const float* __restrict__ Wv,
                                                        const float* __restrict__ bq,
                                                        const float* __restrict__ bk,
                                                        const float* __restrict__ bv,
                                                        _Float16* __restrict__ Wt,
                                                        float* __restrict__ bias) {
  int idx = blockIdx.x * 256 + threadIdx.x;
  if (idx >= DQKV * DM) return;
  int n = idx / DM, k = idx % DM;
  const float* W; const float* b; int nn;
  if (n < DM)          { W = Wq; b = bq; nn = n; }
  else if (n < 2 * DM) { W = Wk; b = bk; nn = n - DM; }
  else                 { W = Wv; b = bv; nn = n - 2 * DM; }
  Wt[idx] = (_Float16)W[(size_t)k * DM + nn];
  if (k == 0) bias[n] = b[nn];
}

__global__ __launch_bounds__(256) void pack_wo_kernel(const float* __restrict__ Wo,
                                                      _Float16* __restrict__ Wt) {
  int idx = blockIdx.x * 256 + threadIdx.x;
  if (idx >= DM * DM) return;
  int n = idx / DM, k = idx % DM;
  Wt[idx] = (_Float16)Wo[(size_t)k * DM + n];
}

// ---------------- WMMA GEMM: C[M,N] = A[M,K] * Bt[N,K]^T + bias ----------------
// 128x128 block tile, K-tiles of 64, 8 waves (4 x 2), 8 accumulators per wave.
// Double-buffered LDS ping-pong: global prefetch of tile k+1 overlaps WMMAs of tile k,
// one barrier per k-tile.

__global__ __launch_bounds__(256) void gemm_wmma_kernel(const _Float16* __restrict__ A,
                                                        const _Float16* __restrict__ Bt,
                                                        const float* __restrict__ bias,
                                                        float* __restrict__ C,
                                                        int K, int N, int Mvalid) {
  __shared__ _Float16 As[2][128 * 64];
  __shared__ _Float16 Bs[2][128 * 64];
  int m0 = blockIdx.x * 128, n0 = blockIdx.y * 128;
  int tid = threadIdx.x, lane = tid & 31, w = tid >> 5;
  int wm = w >> 1, wn = w & 1;
  int half = lane >> 4, c16 = lane & 15;
  // cooperative-load geometry: 4 chunks of 8 halves per thread per matrix
  int lrow = tid >> 3, lcol = (tid & 7) * 8;

  v8f acc[2][4];
#pragma unroll
  for (int i = 0; i < 2; ++i)
#pragma unroll
    for (int j = 0; j < 4; ++j) acc[i][j] = vzero8();

  int nt = K / 64;
  v8h ra[4], rb[4];

  // prefetch tile 0
#pragma unroll
  for (int i = 0; i < 4; ++i) {
    ra[i] = *(const v8h*)&A[(size_t)(m0 + lrow + i * 32) * K + lcol];
    rb[i] = *(const v8h*)&Bt[(size_t)(n0 + lrow + i * 32) * K + lcol];
  }
#pragma unroll
  for (int i = 0; i < 4; ++i) {
    *(v8h*)&As[0][(lrow + i * 32) * 64 + lcol] = ra[i];
    *(v8h*)&Bs[0][(lrow + i * 32) * 64 + lcol] = rb[i];
  }
  __syncthreads();

  for (int kt = 0; kt < nt; ++kt) {
    int cur = kt & 1, nxt = cur ^ 1;
    bool more = (kt + 1) < nt;
    if (more) {
      int k0 = (kt + 1) * 64;
#pragma unroll
      for (int i = 0; i < 4; ++i) {
        ra[i] = *(const v8h*)&A[(size_t)(m0 + lrow + i * 32) * K + k0 + lcol];
        rb[i] = *(const v8h*)&Bt[(size_t)(n0 + lrow + i * 32) * K + k0 + lcol];
      }
    }
    const _Float16* as = As[cur];
    const _Float16* bs = Bs[cur];
#pragma unroll
    for (int ks = 0; ks < 64; ks += 32) {
      // batch all fragment loads, then issue the 8 WMMAs back-to-back
      v16h a0 = lds_frag(as, 64, wm * 32, ks);
      v16h a1 = lds_frag(as, 64, wm * 32 + 16, ks);
      v16h b[4];
#pragma unroll
      for (int j = 0; j < 4; ++j) b[j] = lds_frag(bs, 64, wn * 64 + j * 16, ks);
#pragma unroll
      for (int j = 0; j < 4; ++j) {
        acc[0][j] = wmma_f16(a0, b[j], acc[0][j]);
        acc[1][j] = wmma_f16(a1, b[j], acc[1][j]);
      }
    }
    if (more) {
#pragma unroll
      for (int i = 0; i < 4; ++i) {
        *(v8h*)&As[nxt][(lrow + i * 32) * 64 + lcol] = ra[i];
        *(v8h*)&Bs[nxt][(lrow + i * 32) * 64 + lcol] = rb[i];
      }
    }
    __syncthreads();
  }

#pragma unroll
  for (int mi = 0; mi < 2; ++mi)
#pragma unroll
    for (int nj = 0; nj < 4; ++nj) {
      int colg = n0 + wn * 64 + nj * 16 + c16;
      float bv = bias[colg];
#pragma unroll
      for (int e = 0; e < 8; ++e) {
        int row = m0 + wm * 32 + mi * 16 + e + half * 8;
        if (row < Mvalid) C[(size_t)row * N + colg] = acc[mi][nj][e] + bv;
      }
    }
}

// ---------------- RMSNorm + 3D RoPE + layout to [head][n_pad][128] f16 ----------------

__global__ __launch_bounds__(256) void qkv_post_kernel(const float* __restrict__ Cqkv,
                                                       const float* __restrict__ q_scale,
                                                       const float* __restrict__ k_scale,
                                                       const float* __restrict__ ft,
                                                       const float* __restrict__ fh,
                                                       const float* __restrict__ fw,
                                                       _Float16* __restrict__ Qh,
                                                       _Float16* __restrict__ Kh,
                                                       _Float16* __restrict__ Vh) {
  int row = blockIdx.x, tid = threadIdx.x;
  if (row >= N_TOK) {
    for (int i = tid; i < NH * HD; i += 256) {
      int head = i / HD, d = i % HD;
      size_t o = (size_t)head * (N_PAD * HD) + (size_t)row * HD + d;
      Qh[o] = (_Float16)0.f; Kh[o] = (_Float16)0.f; Vh[o] = (_Float16)0.f;
    }
    return;
  }
  const float* crow = Cqkv + (size_t)row * DQKV;
  int lane = tid & 31, w = tid >> 5;
  __shared__ float redq[8], redk[8], srq, srk;

  float sq = 0.f, sk = 0.f;
  for (int c = tid; c < DM; c += 256) {
    float a = crow[c];      sq += a * a;
    float b = crow[DM + c]; sk += b * b;
  }
#pragma unroll
  for (int o = 16; o; o >>= 1) { sq += __shfl_xor(sq, o, 32); sk += __shfl_xor(sk, o, 32); }
  if (lane == 0) { redq[w] = sq; redk[w] = sk; }
  __syncthreads();
  if (tid == 0) {
    float a = 0.f, b = 0.f;
    for (int i = 0; i < 8; ++i) { a += redq[i]; b += redk[i]; }
    srq = rsqrtf(a / (float)DM + 1e-6f);
    srk = rsqrtf(b / (float)DM + 1e-6f);
  }
  __syncthreads();
  float rqs = srq, rks = srk;

  int t = row / (HHH * WWW), rem = row % (HHH * WWW);
  int y = rem / WWW, xw = rem % WWW;

  for (int p = tid; p < NH * 64; p += 256) {
    int head = p >> 6, j = p & 63;
    int c0 = head * HD + 2 * j;
    float cs, sn;
    if (j < 22)      { const float* fp = ft + ((size_t)t * 22 + j) * 2;         cs = fp[0]; sn = fp[1]; }
    else if (j < 43) { const float* fp = fh + ((size_t)y * 21 + (j - 22)) * 2;  cs = fp[0]; sn = fp[1]; }
    else             { const float* fp = fw + ((size_t)xw * 21 + (j - 43)) * 2; cs = fp[0]; sn = fp[1]; }
    float q0 = crow[c0] * rqs * q_scale[c0];
    float q1 = crow[c0 + 1] * rqs * q_scale[c0 + 1];
    float k0 = crow[DM + c0] * rks * k_scale[c0];
    float k1 = crow[DM + c0 + 1] * rks * k_scale[c0 + 1];
    size_t o = (size_t)head * (N_PAD * HD) + (size_t)row * HD + 2 * j;
    Qh[o]     = (_Float16)(q0 * cs - q1 * sn);
    Qh[o + 1] = (_Float16)(q0 * sn + q1 * cs);
    Kh[o]     = (_Float16)(k0 * cs - k1 * sn);
    Kh[o + 1] = (_Float16)(k0 * sn + k1 * cs);
    Vh[o]     = (_Float16)crow[2 * DM + c0];
    Vh[o + 1] = (_Float16)crow[2 * DM + c0 + 1];
  }
}

// ---------------- Flash attention: one (head, 64-row q tile) per block ----------------

__global__ __launch_bounds__(256) void flash_attn_kernel(const _Float16* __restrict__ Qh,
                                                         const _Float16* __restrict__ Kh,
                                                         const _Float16* __restrict__ Vh,
                                                         _Float16* __restrict__ Ah) {
  int q0 = blockIdx.x * 64;
  int head = blockIdx.y;
  const _Float16* Qg = Qh + (size_t)head * (N_PAD * HD);
  const _Float16* Kg = Kh + (size_t)head * (N_PAD * HD);
  const _Float16* Vg = Vh + (size_t)head * (N_PAD * HD);

  __shared__ _Float16 Qs[64 * 128];
  __shared__ _Float16 Ks[64 * 128];
  __shared__ _Float16 Vt[128 * 64];   // V transposed: [d][j]
  __shared__ _Float16 Ps[64 * 64];
  __shared__ float Mrow[64], Lrow[64], Arow[64];
  __shared__ float pmax[2][64], psum[2][64];

  int tid = threadIdx.x, lane = tid & 31, w = tid >> 5;
  int wm = w >> 1, wn = w & 1;
  int half = lane >> 4, c16 = lane & 15;
  const float scale = 0.08838834764831845f;  // 1/sqrt(128)

  v8f o[4];
#pragma unroll
  for (int j = 0; j < 4; ++j) o[j] = vzero8();

  for (int c = tid; c < 1024; c += 256) {
    int r = c >> 4, cc = (c & 15) * 8;
    *(v8h*)&Qs[r * 128 + cc] = *(const v8h*)&Qg[(size_t)(q0 + r) * HD + cc];
  }
  if (tid < 64) { Mrow[tid] = -3.0e30f; Lrow[tid] = 0.f; }
  __syncthreads();

  for (int j0 = 0; j0 < N_TOK; j0 += 64) {
    // stage K (row-major = N-major B for S) and V transposed (N-major B for O)
    for (int c = tid; c < 1024; c += 256) {
      int r = c >> 4, cc = (c & 15) * 8;
      *(v8h*)&Ks[r * 128 + cc] = *(const v8h*)&Kg[(size_t)(j0 + r) * HD + cc];
      v8h vv = *(const v8h*)&Vg[(size_t)(j0 + r) * HD + cc];
#pragma unroll
      for (int i = 0; i < 8; ++i) Vt[(cc + i) * 64 + r] = vv[i];
    }
    __syncthreads();

    // S = Q K^T  (wave tile: 16 rows x 32 cols)
    v8f s[2];
    s[0] = vzero8(); s[1] = vzero8();
#pragma unroll
    for (int ks = 0; ks < 128; ks += 32) {
      v16h aq = lds_frag(Qs, 128, wm * 16, ks);
      v16h b0 = lds_frag(Ks, 128, wn * 32, ks);
      v16h b1 = lds_frag(Ks, 128, wn * 32 + 16, ks);
      s[0] = wmma_f16(aq, b0, s[0]);
      s[1] = wmma_f16(aq, b1, s[1]);
    }

    // scale + mask + per-row max
    bool msk0 = (j0 + wn * 32 + c16) >= N_TOK;
    bool msk1 = (j0 + wn * 32 + 16 + c16) >= N_TOK;
    float rmax[8];
#pragma unroll
    for (int e = 0; e < 8; ++e) {
      s[0][e] = msk0 ? -1.0e30f : s[0][e] * scale;
      s[1][e] = msk1 ? -1.0e30f : s[1][e] * scale;
      float v = fmaxf(s[0][e], s[1][e]);
#pragma unroll
      for (int d = 8; d; d >>= 1) v = fmaxf(v, __shfl_xor(v, d, 16));
      rmax[e] = v;
    }
    if (c16 == 0) {
#pragma unroll
      for (int e = 0; e < 8; ++e) pmax[wn][wm * 16 + e + half * 8] = rmax[e];
    }
    __syncthreads();

    if (tid < 64) {
      float mn = fmaxf(Mrow[tid], fmaxf(pmax[0][tid], pmax[1][tid]));
      Arow[tid] = __expf(Mrow[tid] - mn);
      Mrow[tid] = mn;
    }
    __syncthreads();

    // P = exp(S - m), stage to LDS, partial row sums, rescale O
    float mrow_e[8], al_e[8], rsum[8];
#pragma unroll
    for (int e = 0; e < 8; ++e) {
      int rl = wm * 16 + e + half * 8;
      mrow_e[e] = Mrow[rl];
      al_e[e] = Arow[rl];
    }
#pragma unroll
    for (int e = 0; e < 8; ++e) {
      int rl = wm * 16 + e + half * 8;
      float p0 = __expf(s[0][e] - mrow_e[e]);
      float p1 = __expf(s[1][e] - mrow_e[e]);
      Ps[rl * 64 + wn * 32 + c16]      = (_Float16)p0;
      Ps[rl * 64 + wn * 32 + 16 + c16] = (_Float16)p1;
      float v = p0 + p1;
#pragma unroll
      for (int d = 8; d; d >>= 1) v += __shfl_xor(v, d, 16);
      rsum[e] = v;
    }
    if (c16 == 0) {
#pragma unroll
      for (int e = 0; e < 8; ++e) psum[wn][wm * 16 + e + half * 8] = rsum[e];
    }
#pragma unroll
    for (int nj = 0; nj < 4; ++nj)
#pragma unroll
      for (int e = 0; e < 8; ++e) o[nj][e] *= al_e[e];
    __syncthreads();

    if (tid < 64) Lrow[tid] = Lrow[tid] * Arow[tid] + psum[0][tid] + psum[1][tid];

    // O += P * V   (wave tile: 16 rows x 64 d-cols); batch fragments, then WMMAs
#pragma unroll
    for (int ks = 0; ks < 64; ks += 32) {
      v16h ap = lds_frag(Ps, 64, wm * 16, ks);
      v16h bv[4];
#pragma unroll
      for (int nj = 0; nj < 4; ++nj) bv[nj] = lds_frag(Vt, 64, wn * 64 + nj * 16, ks);
#pragma unroll
      for (int nj = 0; nj < 4; ++nj) o[nj] = wmma_f16(ap, bv[nj], o[nj]);
    }
    __syncthreads();
  }

  float linv[8];
#pragma unroll
  for (int e = 0; e < 8; ++e) {
    float L = Lrow[wm * 16 + e + half * 8];
    linv[e] = (L > 0.f) ? 1.f / L : 0.f;
  }
#pragma unroll
  for (int nj = 0; nj < 4; ++nj)
#pragma unroll
    for (int e = 0; e < 8; ++e) {
      int rl = wm * 16 + e + half * 8;
      int grow = q0 + rl;
      int d = wn * 64 + nj * 16 + c16;
      _Float16 val = (grow < N_TOK) ? (_Float16)(o[nj][e] * linv[e]) : (_Float16)0.f;
      Ah[(size_t)grow * DM + head * HD + d] = val;
    }
}

// ---------------- host launch ----------------

extern "C" void kernel_launch(void* const* d_in, const int* in_sizes, int n_in,
                              void* d_out, int out_size, void* d_ws, size_t ws_size,
                              hipStream_t stream) {
  const float* x  = (const float*)d_in[0];
  const float* Wq = (const float*)d_in[1];
  const float* bq = (const float*)d_in[2];
  const float* Wk = (const float*)d_in[3];
  const float* bk = (const float*)d_in[4];
  const float* Wv = (const float*)d_in[5];
  const float* bv = (const float*)d_in[6];
  const float* Wo = (const float*)d_in[7];
  const float* bo = (const float*)d_in[8];
  const float* q_scale = (const float*)d_in[9];
  const float* k_scale = (const float*)d_in[10];
  const float* ft = (const float*)d_in[11];
  const float* fh = (const float*)d_in[12];
  const float* fw = (const float*)d_in[13];
  float* out = (float*)d_out;

  char* wsb = (char*)d_ws;
  size_t off = 0;
  _Float16* xh    = (_Float16*)(wsb + off); off += (size_t)N_PAD * DM * 2;       // 9,043,968
  _Float16* Wtqkv = (_Float16*)(wsb + off); off += (size_t)DQKV * DM * 2;        // 14,155,776
  float*    bqkv  = (float*)   (wsb + off); off += (size_t)DQKV * 4;             // 18,432
  _Float16* Wto   = (_Float16*)(wsb + off); off += (size_t)DM * DM * 2;          // 4,718,592
  float*    Cqkv  = (float*)   (wsb + off); off += (size_t)N_PAD * DQKV * 4;     // 54,263,808
  _Float16* Qhb   = (_Float16*)(wsb + off); off += (size_t)NH * N_PAD * HD * 2;  // 9,043,968
  _Float16* Khb   = (_Float16*)(wsb + off); off += (size_t)NH * N_PAD * HD * 2;
  _Float16* Vhb   = (_Float16*)(wsb + off); off += (size_t)NH * N_PAD * HD * 2;
  _Float16* Ahb   = (_Float16*)(wsb + off); off += (size_t)N_PAD * DM * 2;

  pack_x_kernel<<<(N_PAD * DM + 255) / 256, 256, 0, stream>>>(x, xh);
  pack_wqkv_kernel<<<(DQKV * DM + 255) / 256, 256, 0, stream>>>(Wq, Wk, Wv, bq, bk, bv, Wtqkv, bqkv);
  pack_wo_kernel<<<(DM * DM + 255) / 256, 256, 0, stream>>>(Wo, Wto);

  // QKV projection: [2944 x 1536] x [1536 x 4608] -> f32
  gemm_wmma_kernel<<<dim3(N_PAD / 128, DQKV / 128), 256, 0, stream>>>(
      xh, Wtqkv, bqkv, Cqkv, DM, DQKV, N_PAD);

  qkv_post_kernel<<<N_PAD, 256, 0, stream>>>(Cqkv, q_scale, k_scale, ft, fh, fw, Qhb, Khb, Vhb);

  flash_attn_kernel<<<dim3(N_PAD / 64, NH), 256, 0, stream>>>(Qhb, Khb, Vhb, Ahb);

  // output projection: [2944 x 1536] x [1536 x 1536] -> f32, clamp rows to 2925
  gemm_wmma_kernel<<<dim3(N_PAD / 128, DM / 128), 256, 0, stream>>>(
      Ahb, Wto, bo, out, DM, DM, N_TOK);
}
